// FusedMoE_19567871001237
// MI455X (gfx1250) — matrix operand order repaired and checked by reading, not compiled, not run on hardware
//
#include <hip/hip_runtime.h>
#include <hip/hip_bf16.h>
#include <stdint.h>

// ---------------- problem constants ----------------
#define NUM_EXPERTS 32
#define HIDDEN      2048
#define INTER       2816
#define HALF_INTER  1408
#define TOKENS      1024
#define TOPK        8
#define CAP         512
#define NASSIGN     (TOKENS * TOPK)

// ---------------- CDNA5 WMMA types ----------------
typedef __bf16 bf16;
typedef __attribute__((ext_vector_type(16))) __bf16 v16bf;
typedef __attribute__((ext_vector_type(8)))  float  v8f;

__device__ __forceinline__ bf16 f2bf(float f) {
  uint32_t u = __builtin_bit_cast(uint32_t, f);
  uint32_t r = u + 0x7FFFu + ((u >> 16) & 1u);   // round-to-nearest-even
  uint16_t h = (uint16_t)(r >> 16);
  return __builtin_bit_cast(bf16, h);
}

// LDS row stride for K=64 stages: 64 data + 8 pad bf16 -> 144B rows
// (16B aligned for b128 staging, rotates banks by 36 words between rows).
#define LDA 72

// ---- A fragment (16x32 bf16, MxK), ISA 7.12.2 layout:
// lanes 0-15: M=lane,   elems0-7 -> K=0..7,  elems8-15 -> K=16..23
// lanes16-31: M=lane-16,elems0-7 -> K=8..15, elems8-15 -> K=24..31
__device__ __forceinline__ v16bf load_frag_a(const bf16* sA, int m0, int kk, int lane) {
  const int r  = lane & 15;
  const int kh = (lane >> 4) * 8;
  const bf16* p = sA + (m0 + r) * LDA + kk;
  v16bf f;
#pragma unroll
  for (int q = 0; q < 8; ++q) f[q] = p[kh + q];
#pragma unroll
  for (int q = 0; q < 8; ++q) f[8 + q] = p[16 + kh + q];
  return f;
}

// ---- B fragment (32x16 bf16, KxN): column N = lane%16,
// lanes 0-15 hold K=0..7,16..23; lanes 16-31 hold K=8..15,24..31 (matches A).
__device__ __forceinline__ v16bf load_frag_b(const bf16* sB, int n0, int kk, int lane) {
  const int c  = lane & 15;
  const int kb = (lane >> 4) * 8;
  const bf16* p = sB + (n0 + c) * LDA + kk + kb;
  v16bf f;
#pragma unroll
  for (int q = 0; q < 16; ++q) f[q] = p[q < 8 ? q : q + 8];
  return f;
}

#define WMMA_BF16(A, B, C) \
  __builtin_amdgcn_wmma_f32_16x16x32_bf16(false, (A), false, (B), (short)0, (C), false, false)

// ---------------- routing ----------------
__global__ void k_zero(bf16* __restrict__ xb, int* __restrict__ counts) {
  size_t i = (size_t)blockIdx.x * blockDim.x + threadIdx.x;
  const size_t n16 = (size_t)NUM_EXPERTS * CAP * HIDDEN * 2 / 16;
  if (i < n16) reinterpret_cast<uint4*>(xb)[i] = make_uint4(0u, 0u, 0u, 0u);
  if (i < NUM_EXPERTS) counts[i] = 0;
}

// Deterministic stable rank within expert (matches stable argsort semantics).
// LDS-tiled O(n^2/block) scan; counts via order-independent atomicAdd.
__global__ void k_route(const int* __restrict__ topk_idx,
                        int* __restrict__ pos, int* __restrict__ counts) {
  __shared__ int cache[256];
  const int i = blockIdx.x * 256 + threadIdx.x;       // grid covers NASSIGN exactly
  const int e = topk_idx[i];
  const int limit = blockIdx.x * 256 + 256;           // block-uniform chunk bound
  int p = 0;
  for (int c0 = 0; c0 < limit; c0 += 256) {
    __syncthreads();
    cache[threadIdx.x] = topk_idx[c0 + threadIdx.x];
    __syncthreads();
    const int jmax = i - c0 < 256 ? (i - c0) : 256;
    for (int j = 0; j < jmax; ++j) p += (cache[j] == e) ? 1 : 0;
  }
  pos[i] = p;
  atomicAdd(&counts[e], 1);
}

__global__ void k_scatter(const float* __restrict__ hs,
                          const int* __restrict__ topk_idx,
                          const int* __restrict__ pos,
                          bf16* __restrict__ xb) {
  const int a = blockIdx.x;           // assignment id
  const int t = a / TOPK;
  const int e = topk_idx[a];
  const int p = pos[a];
  if (p >= CAP) return;               // capacity drop
  const float* src = hs + (size_t)t * HIDDEN;
  bf16* dst = xb + ((size_t)e * CAP + p) * HIDDEN;
  for (int h = threadIdx.x; h < HIDDEN; h += blockDim.x) dst[h] = f2bf(src[h]);
}

// ---- cooperative staging helpers (256 threads) ----
// A tile: 128 rows x 64 bf16.  2 threads/row, 32 bf16 (64B) each -> b128 x4.
__device__ __forceinline__ void stage_a(bf16* __restrict__ sA,
                                        const bf16* __restrict__ X,
                                        int ldx, int k0, int tid) {
  const int row = tid >> 1;
  const int col = (tid & 1) * 32;
  const uint4* s = reinterpret_cast<const uint4*>(X + (size_t)row * ldx + k0 + col);
  uint4* d = reinterpret_cast<uint4*>(sA + row * LDA + col);
#pragma unroll
  for (int q = 0; q < 4; ++q) d[q] = s[q];
}

// B tile: 64 rows x 64 f32 -> cvt bf16.  4 threads/row, 16 f32 each.
__device__ __forceinline__ void stage_b(bf16* __restrict__ sB,
                                        const float* __restrict__ W,
                                        int ldw, int k0, int tid) {
  const int row = tid >> 2;
  const int col = (tid & 3) * 16;
  const float4* s = reinterpret_cast<const float4*>(W + (size_t)row * ldw + k0 + col);
  bf16 tmp[16];
#pragma unroll
  for (int q = 0; q < 4; ++q) {
    const float4 v = s[q];
    tmp[4 * q + 0] = f2bf(v.x);
    tmp[4 * q + 1] = f2bf(v.y);
    tmp[4 * q + 2] = f2bf(v.z);
    tmp[4 * q + 3] = f2bf(v.w);
  }
  uint4* d = reinterpret_cast<uint4*>(sB + row * LDA + col);
  d[0] = reinterpret_cast<const uint4*>(tmp)[0];
  d[1] = reinterpret_cast<const uint4*>(tmp)[1];
}

// ---------------- GEMM1 + SiLU*up fusion ----------------
// Block tile: M=128 x (64 gate + 64 up) cols, K-stage 64. 8 waves in 4(M) x 2(N).
__global__ __launch_bounds__(256) void k_gemm1(const bf16* __restrict__ xb,
                                               const float* __restrict__ w1,
                                               const int* __restrict__ counts,
                                               bf16* __restrict__ ab) {
  const int NT = HALF_INTER / 64;               // 22
  const int nt = blockIdx.x % NT;
  const int mt = (blockIdx.x / NT) % (CAP / 128);
  const int e  = blockIdx.x / (NT * (CAP / 128));
  int cnt = counts[e]; if (cnt > CAP) cnt = CAP;
  if (mt * 128 >= cnt) return;                  // skip empty capacity tiles

  __shared__ __align__(16) bf16 sA [128 * LDA];   // 18 KB
  __shared__ __align__(16) bf16 sBg[ 64 * LDA];   //  9 KB
  __shared__ __align__(16) bf16 sBu[ 64 * LDA];   //  9 KB

  const int tid  = threadIdx.x;
  const int lane = tid & 31;
  const int wave = tid >> 5;
  const int wm   = wave >> 1;                   // 0..3
  const int wn   = wave & 1;                    // 0..1

  const bf16*  Xe = xb + ((size_t)e * CAP + mt * 128) * HIDDEN;
  const float* Wg = w1 + (size_t)e * INTER * HIDDEN + (size_t)(nt * 64) * HIDDEN;
  const float* Wu = Wg + (size_t)HALF_INTER * HIDDEN;

  const v8f vz = {0.f, 0.f, 0.f, 0.f, 0.f, 0.f, 0.f, 0.f};
  v8f cg[2][2], cu[2][2];
#pragma unroll
  for (int i = 0; i < 2; ++i)
#pragma unroll
    for (int j = 0; j < 2; ++j) { cg[i][j] = vz; cu[i][j] = vz; }

  for (int k0 = 0; k0 < HIDDEN; k0 += 64) {
    __syncthreads();
    stage_a(sA, Xe, HIDDEN, k0, tid);
    stage_b(sBg, Wg, HIDDEN, k0, tid);
    stage_b(sBu, Wu, HIDDEN, k0, tid);
    if (k0 + 64 < HIDDEN) {   // prefetch next weight stage -> global_prefetch_b8
      __builtin_prefetch(Wg + (size_t)(tid >> 2) * HIDDEN + k0 + 64 + (tid & 3) * 16, 0, 1);
      __builtin_prefetch(Wu + (size_t)(tid >> 2) * HIDDEN + k0 + 64 + (tid & 3) * 16, 0, 1);
    }
    __syncthreads();

#pragma unroll
    for (int kk = 0; kk < 64; kk += 32) {
      const v16bf a0  = load_frag_a(sA,  wm * 32,      kk, lane);
      const v16bf a1  = load_frag_a(sA,  wm * 32 + 16, kk, lane);
      const v16bf bg0 = load_frag_b(sBg, wn * 32,      kk, lane);
      const v16bf bg1 = load_frag_b(sBg, wn * 32 + 16, kk, lane);
      const v16bf bu0 = load_frag_b(sBu, wn * 32,      kk, lane);
      const v16bf bu1 = load_frag_b(sBu, wn * 32 + 16, kk, lane);

      cg[0][0] = WMMA_BF16(a0, bg0, cg[0][0]);
      cg[0][1] = WMMA_BF16(a0, bg1, cg[0][1]);
      cg[1][0] = WMMA_BF16(a1, bg0, cg[1][0]);
      cg[1][1] = WMMA_BF16(a1, bg1, cg[1][1]);
      cu[0][0] = WMMA_BF16(a0, bu0, cu[0][0]);
      cu[0][1] = WMMA_BF16(a0, bu1, cu[0][1]);
      cu[1][0] = WMMA_BF16(a1, bu0, cu[1][0]);
      cu[1][1] = WMMA_BF16(a1, bu1, cu[1][1]);
    }
  }

  // epilogue: a = silu(gate) * up  (fast rcp; inputs already bf16-truncated)
  // C/D layout: lanes 0-15: VGPR r -> M=r; lanes 16-31: VGPR r -> M=r+8; N=lane%16
  const int mhi  = (lane >> 4) * 8;
  const int ncol = lane & 15;
  bf16* Ab = ab + (size_t)e * CAP * HALF_INTER;
#pragma unroll
  for (int mi = 0; mi < 2; ++mi)
#pragma unroll
    for (int ni = 0; ni < 2; ++ni)
#pragma unroll
      for (int r = 0; r < 8; ++r) {
        const float g = cg[mi][ni][r];
        const float u = cu[mi][ni][r];
        const float v = g * __builtin_amdgcn_rcpf(1.0f + __expf(-g)) * u;
        const int row = mt * 128 + wm * 32 + mi * 16 + mhi + r;
        const int col = nt * 64 + wn * 32 + ni * 16 + ncol;
        Ab[(size_t)row * HALF_INTER + col] = f2bf(v);
      }
}

// ---------------- GEMM2 ----------------
// Block tile: M=128 x N=64, K=1408 (22 stages of 64).
__global__ __launch_bounds__(256) void k_gemm2(const bf16* __restrict__ ab,
                                               const float* __restrict__ w2,
                                               const int* __restrict__ counts,
                                               float* __restrict__ ob) {
  const int NT = HIDDEN / 64;                   // 32
  const int nt = blockIdx.x % NT;
  const int mt = (blockIdx.x / NT) % (CAP / 128);
  const int e  = blockIdx.x / (NT * (CAP / 128));
  int cnt = counts[e]; if (cnt > CAP) cnt = CAP;
  if (mt * 128 >= cnt) return;

  __shared__ __align__(16) bf16 sA[128 * LDA];
  __shared__ __align__(16) bf16 sB[ 64 * LDA];

  const int tid  = threadIdx.x;
  const int lane = tid & 31;
  const int wave = tid >> 5;
  const int wm   = wave >> 1;
  const int wn   = wave & 1;

  const bf16*  Xe = ab + ((size_t)e * CAP + mt * 128) * HALF_INTER;
  const float* W  = w2 + (size_t)e * HIDDEN * HALF_INTER + (size_t)(nt * 64) * HALF_INTER;

  const v8f vz = {0.f, 0.f, 0.f, 0.f, 0.f, 0.f, 0.f, 0.f};
  v8f c[2][2];
#pragma unroll
  for (int i = 0; i < 2; ++i)
#pragma unroll
    for (int j = 0; j < 2; ++j) c[i][j] = vz;

  for (int k0 = 0; k0 < HALF_INTER; k0 += 64) {
    __syncthreads();
    stage_a(sA, Xe, HALF_INTER, k0, tid);
    stage_b(sB, W, HALF_INTER, k0, tid);
    if (k0 + 64 < HALF_INTER)
      __builtin_prefetch(W + (size_t)(tid >> 2) * HALF_INTER + k0 + 64 + (tid & 3) * 16, 0, 1);
    __syncthreads();

#pragma unroll
    for (int kk = 0; kk < 64; kk += 32) {
      const v16bf a0 = load_frag_a(sA, wm * 32,      kk, lane);
      const v16bf a1 = load_frag_a(sA, wm * 32 + 16, kk, lane);
      const v16bf b0 = load_frag_b(sB, wn * 32,      kk, lane);
      const v16bf b1 = load_frag_b(sB, wn * 32 + 16, kk, lane);

      c[0][0] = WMMA_BF16(a0, b0, c[0][0]);
      c[0][1] = WMMA_BF16(a0, b1, c[0][1]);
      c[1][0] = WMMA_BF16(a1, b0, c[1][0]);
      c[1][1] = WMMA_BF16(a1, b1, c[1][1]);
    }
  }

  const int mhi  = (lane >> 4) * 8;
  const int ncol = lane & 15;
  float* Ob = ob + (size_t)e * CAP * HIDDEN;
#pragma unroll
  for (int mi = 0; mi < 2; ++mi)
#pragma unroll
    for (int ni = 0; ni < 2; ++ni)
#pragma unroll
      for (int r = 0; r < 8; ++r) {
        const int row = mt * 128 + wm * 32 + mi * 16 + mhi + r;
        const int col = nt * 64 + wn * 32 + ni * 16 + ncol;
        Ob[(size_t)row * HIDDEN + col] = c[mi][ni][r];
      }
}

// ---------------- combine (deterministic per-token sum) ----------------
__global__ void k_combine(const float* __restrict__ ob,
                          const int* __restrict__ topk_idx,
                          const float* __restrict__ topk_w,
                          const int* __restrict__ pos,
                          float* __restrict__ out) {
  const int t = blockIdx.x;
  for (int h = threadIdx.x; h < HIDDEN; h += blockDim.x) {
    float acc = 0.f;
#pragma unroll
    for (int k = 0; k < TOPK; ++k) {
      const int a = t * TOPK + k;
      const int p = pos[a];
      if (p < CAP) {
        const int e = topk_idx[a];
        acc += topk_w[a] * ob[((size_t)e * CAP + p) * HIDDEN + h];
      }
    }
    out[(size_t)t * HIDDEN + h] = acc;
  }
}

// ---------------- launcher ----------------
extern "C" void kernel_launch(void* const* d_in, const int* in_sizes, int n_in,
                              void* d_out, int out_size, void* d_ws, size_t ws_size,
                              hipStream_t stream) {
  const float* hs = (const float*)d_in[0];   // hidden_states [1024,2048] f32
  const float* tw = (const float*)d_in[1];   // topk_weights  [1024,8]    f32
  const float* w1 = (const float*)d_in[2];   // w1 [32,2816,2048] f32
  const float* w2 = (const float*)d_in[3];   // w2 [32,2048,1408] f32
  const int*   ti = (const int*)  d_in[4];   // topk_idx [1024,8] i32

  char* ws = (char*)d_ws;
  int*   counts = (int*)(ws + 0);                         // 32 ints
  int*   pos    = (int*)(ws + 256);                       // 8192 ints
  const size_t XB_OFF  = 65536;
  const size_t XB_SZ   = (size_t)NUM_EXPERTS * CAP * HIDDEN * 2;       // 64 MB
  const size_t AB_OFF  = XB_OFF + XB_SZ;
  const size_t AB_SZ   = (size_t)NUM_EXPERTS * CAP * HALF_INTER * 2;   // 46 MB
  const size_t OB_OFF  = AB_OFF + AB_SZ;
  bf16*  xb = (bf16*)(ws + XB_OFF);
  bf16*  ab = (bf16*)(ws + AB_OFF);
  float* ob = (float*)(ws + OB_OFF);                                   // 128 MB
  float* out = (float*)d_out;

  // 1. zero capacity buffer + expert counts
  k_zero<<<dim3(16384), dim3(256), 0, stream>>>(xb, counts);
  // 2. deterministic routing (stable per-expert ranks)
  k_route<<<dim3(NASSIGN / 256), dim3(256), 0, stream>>>(ti, pos, counts);
  // 3. gather tokens into per-expert bf16 buffers
  k_scatter<<<dim3(NASSIGN), dim3(256), 0, stream>>>(hs, ti, pos, xb);
  // 4. GEMM1 + SiLU*up  (32 experts x 4 m-tiles x 22 n-tiles)
  k_gemm1<<<dim3(NUM_EXPERTS * (CAP / 128) * (HALF_INTER / 64)), dim3(256), 0, stream>>>(
      xb, w1, counts, ab);
  // 5. GEMM2            (32 experts x 4 m-tiles x 32 n-tiles)
  k_gemm2<<<dim3(NUM_EXPERTS * (CAP / 128) * (HIDDEN / 64)), dim3(256), 0, stream>>>(
      ab, w2, counts, ob);
  // 6. weighted combine back to tokens
  k_combine<<<dim3(TOKENS), dim3(256), 0, stream>>>(ob, ti, tw, pos, out);
}
